// TransstnSkipBlock_57466662420732
// MI455X (gfx1250) — matrix operand order repaired and checked by reading, not compiled, not run on hardware
//
#include <hip/hip_runtime.h>
#include <hip/hip_bf16.h>

typedef __attribute__((ext_vector_type(16))) _Float16 v16h;
typedef __attribute__((ext_vector_type(8)))  _Float16 v8h;
typedef __attribute__((ext_vector_type(8)))  float    v8f;

union Frag16 { v16h v; v8h h[2]; };
union H8     { v8h v; _Float16 h[8]; };
struct F2    { float a, b; };  // alignof==4: safe paired-corner load at any float index

#define C_CH   256
#define KTOT   2304      // 9 * 256, kk-major: k = kk*256 + c
#define TILE_M 32        // pixels per workgroup (one half W-row)

// ---------------------------------------------------------------------------
// Kernel 1: fp32 -> f16 weight repack with K-permutation:
//   wh[o][kk*256 + c] = w[(o,c,kk)]   (GEMM sum is K-order independent)
// ---------------------------------------------------------------------------
__global__ __launch_bounds__(256) void convert_w_kernel(const float* __restrict__ w,
                                                        _Float16* __restrict__ wh,
                                                        int n) {
    int i = blockIdx.x * 256 + threadIdx.x;
    if (i < n) {
        int o  = i / KTOT;
        int r  = i - o * KTOT;
        int kk = r >> 8;           // 0..8
        int c  = r & 255;
        wh[i] = (_Float16)w[o * KTOT + c * 9 + kk];
    }
}

// ---------------------------------------------------------------------------
// Kernel 2: offset conv (6 channels, 3x3 over 256 ch) + build_offset.
// One thread per output pixel; weights staged in LDS (13824 floats = 55KB).
// ---------------------------------------------------------------------------
__global__ __launch_bounds__(256) void offsets_kernel(const float* __restrict__ x,
                                                      const float* __restrict__ w_tm,
                                                      const float* __restrict__ b_tm,
                                                      const float* __restrict__ w_tr,
                                                      const float* __restrict__ b_tr,
                                                      float* __restrict__ dyo,
                                                      float* __restrict__ dxo,
                                                      int H, int W) {
    const int HW = H * W;
    __shared__ float s_wt[KTOT * 6];   // [(c*9+kk)*6 + j]
    int tid = threadIdx.x;
    for (int i = tid; i < KTOT * 6; i += 256) {
        int ck = i / 6, j = i % 6;
        s_wt[i] = (j < 4) ? w_tm[j * KTOT + ck] : w_tr[(j - 4) * KTOT + ck];
    }
    __syncthreads();

    int p = blockIdx.x * 256 + tid;     // pixel id over N*H*W
    int n = p / HW;
    int hw = p % HW;
    int h = hw / W, w = hw % W;

    float acc[6];
    acc[0] = b_tm[0]; acc[1] = b_tm[1]; acc[2] = b_tm[2]; acc[3] = b_tm[3];
    acc[4] = b_tr[0]; acc[5] = b_tr[1];

    const float* xb = x + (size_t)n * C_CH * HW;
    for (int c = 0; c < C_CH; ++c) {
        const float* xc = xb + c * HW;
        const float* wp = &s_wt[c * 54];
#pragma unroll
        for (int kk = 0; kk < 9; ++kk) {
            int yy = h - 1 + kk / 3;
            int xx = w - 1 + kk % 3;
            float v = 0.f;
            if ((unsigned)yy < (unsigned)H && (unsigned)xx < (unsigned)W)
                v = xc[yy * W + xx];
#pragma unroll
            for (int j = 0; j < 6; ++j) acc[j] = fmaf(wp[kk * 6 + j], v, acc[j]);
        }
    }

    const float RY[9] = {-1,-1,-1, 0,0,0, 1,1,1};
    const float RX[9] = {-1, 0, 1,-1,0,1,-1,0,1};
    float m00 = acc[0], m01 = acc[1], m10 = acc[2], m11 = acc[3];
    float t0 = acc[4], t1 = acc[5];
#pragma unroll
    for (int k = 0; k < 9; ++k) {
        float dy = m00 * RY[k] + m01 * RX[k] - RY[k] + t0;
        float dx = m10 * RY[k] + m11 * RX[k] - RX[k] + t1;
        dyo[((size_t)n * 9 + k) * HW + hw] = dy;
        dxo[((size_t)n * 9 + k) * HW + hw] = dx;
    }
}

// ---------------------------------------------------------------------------
// Kernel 3: deform-conv as implicit GEMM via v_wmma_f32_16x16x32_f16.
// WG = 128 threads (4 waves). Tile: M=32 pixels x 256 out-channels.
// K is kk-major: each 32-wide K-chunk has one fixed 3x3 tap and 32 channels,
// so per thread per K-step: 1 corner-table read, 16 paired b64 gathers along
// channel planes, 1 ds_store_b128 into the A tile, then 8 WMMAs per wave.
// ---------------------------------------------------------------------------
__global__ __launch_bounds__(128) void deform_gemm_kernel(const float* __restrict__ x,
                                                          const float* __restrict__ dy,
                                                          const float* __restrict__ dx,
                                                          const _Float16* __restrict__ wh, // [256][2304] kk-major K
                                                          float* __restrict__ out,
                                                          int H, int W) {
    const int HW = H * W;
    int p0 = blockIdx.x * TILE_M;    // 32 consecutive pixels, same (n,h) since W%32==0
    int n = p0 / HW;
    int hw0 = p0 % HW;
    int h = hw0 / W, w0 = hw0 % W;

    int tid  = threadIdx.x;
    int lane = tid & 31;
    int wave = tid >> 5;

    __shared__ __align__(16) int      s_iA[TILE_M * 9];      // row0 pair base index
    __shared__ __align__(16) int      s_iB[TILE_M * 9];      // row1 pair base index
    __shared__ __align__(16) float4   s_wq[TILE_M * 9];      // 4 redistributed weights
    __shared__ __align__(16) _Float16 s_a[TILE_M][32];       // A tile (32x32 f16)

    // Precompute paired bilinear corners for 32 pixels x 9 taps.
    for (int e = tid; e < TILE_M * 9; e += 128) {
        int m = e / 9, kk = e % 9;
        int w = w0 + m;
        float py = (float)(h - 1 + kk / 3) + dy[((size_t)n * 9 + kk) * HW + h * W + w];
        float px = (float)(w - 1 + kk % 3) + dx[((size_t)n * 9 + kk) * HW + h * W + w];
        float y0f = floorf(py), x0f = floorf(px);
        int   y0 = (int)y0f,    x0 = (int)x0f;
        float wy1 = py - y0f, wy0 = 1.f - wy1;
        float wx1 = px - x0f, wx0 = 1.f - wx1;

        // clamp pair base so [x0c, x0c+1] is always in-bounds, then place the
        // x-weights onto whichever element corresponds to a valid corner.
        int x0c = x0 < 0 ? 0 : (x0 > W - 2 ? W - 2 : x0);
        float ex0 = 0.f, ex1 = 0.f;
#pragma unroll
        for (int j = 0; j < 2; ++j) {
            int rel = x0c + j - x0;       // which corner this element is
            float ew = 0.f;
            if (rel == 0 && x0 >= 0 && x0 < W)              ew = wx0;
            else if (rel == 1 && x0 + 1 >= 0 && x0 + 1 < W) ew = wx1;
            if (j == 0) ex0 = ew; else ex1 = ew;
        }
        bool v0 = (y0 >= 0) && (y0 < H);
        bool v1 = (y0 + 1 >= 0) && (y0 + 1 < H);
        int y0c = y0 < 0 ? 0 : (y0 > H - 1 ? H - 1 : y0);
        int y1c = y0 + 1 < 0 ? 0 : (y0 + 1 > H - 1 ? H - 1 : y0 + 1);
        s_iA[e] = y0c * W + x0c;
        s_iB[e] = y1c * W + x0c;
        s_wq[e] = make_float4(v0 ? wy0 * ex0 : 0.f, v0 ? wy0 * ex1 : 0.f,
                              v1 ? wy1 * ex0 : 0.f, v1 ? wy1 * ex1 : 0.f);
    }

    v8f acc[2][4];
#pragma unroll
    for (int s = 0; s < 2; ++s)
#pragma unroll
        for (int t = 0; t < 4; ++t) acc[s][t] = v8f{};

    const int oc0  = wave * 64;          // 4 tiles of 16 out-channels per wave
    const int half = lane >> 4;          // K half-select for 16-bit fragments
    const int am   = lane & 15;          // row/column within a 16-wide tile
    const int mth  = (tid * 8) >> 5;     // this thread's A row (0..31)
    const int klb  = (tid * 8) & 31;     // base kl within chunk (0,8,16,24)
    const float* xn = x + (size_t)n * C_CH * HW;

    for (int k0 = 0; k0 < KTOT; k0 += 32) {
        __syncthreads();                 // protect s_a against previous reads

        // Hoist B fragments: independent of LDS, in flight during sampling.
        Frag16 bf[4];
#pragma unroll
        for (int t = 0; t < 4; ++t) {
            const _Float16* bp = wh + (size_t)(oc0 + t * 16 + am) * KTOT + k0 + half * 8;
            bf[t].h[0] = *(const v8h*)bp;
            bf[t].h[1] = *(const v8h*)(bp + 16);
        }

        // A-tile fill: fixed tap kk0, 8 consecutive channels per thread.
        int kk0 = k0 >> 8;               // 0..8
        int c0  = k0 & 255;              // multiple of 32
        int e   = mth * 9 + kk0;
        int iA  = s_iA[e];
        int iB  = s_iB[e];
        float4 wq = s_wq[e];
        const float* xb0 = xn + (size_t)(c0 + klb) * HW;
        H8 o8;
#pragma unroll
        for (int j = 0; j < 8; ++j) {
            const float* xc = xb0 + (size_t)j * HW;
            F2 ra = *(const F2*)(xc + iA);
            F2 rb = *(const F2*)(xc + iB);
            o8.h[j] = (_Float16)(wq.x * ra.a + wq.y * ra.b + wq.z * rb.a + wq.w * rb.b);
        }
        *(v8h*)&s_a[mth][klb] = o8.v;
        if (k0 + 32 < KTOT) {
            // warm next K-chunk's first channel plane (global_prefetch_b8)
            int k1 = k0 + 32;
            __builtin_prefetch(xn + (size_t)((k1 & 255) + klb) * HW + iA, 0, 1);
        }
        __syncthreads();

        // A fragments for both M-subtiles (ISA 16-bit A 16x32 layout).
        Frag16 af0, af1;
        af0.h[0] = *(const v8h*)&s_a[am][half * 8];
        af0.h[1] = *(const v8h*)&s_a[am][16 + half * 8];
        af1.h[0] = *(const v8h*)&s_a[16 + am][half * 8];
        af1.h[1] = *(const v8h*)&s_a[16 + am][16 + half * 8];

#pragma unroll
        for (int t = 0; t < 4; ++t) {
            acc[0][t] = __builtin_amdgcn_wmma_f32_16x16x32_f16(
                false, af0.v, false, bf[t].v, (short)0, acc[0][t], false, false);
            acc[1][t] = __builtin_amdgcn_wmma_f32_16x16x32_f16(
                false, af1.v, false, bf[t].v, (short)0, acc[1][t], false, false);
        }
    }

    // Store D: VGPR r -> row m = r + 8*(lane>>4), col = lane&15.
    float* outn = out + (size_t)n * C_CH * HW + h * W + w0;
    int mbase = half * 8;
#pragma unroll
    for (int s = 0; s < 2; ++s) {
#pragma unroll
        for (int t = 0; t < 4; ++t) {
            int oc = oc0 + t * 16 + am;
#pragma unroll
            for (int r = 0; r < 8; ++r) {
                outn[(size_t)oc * HW + s * 16 + mbase + r] = acc[s][t][r];
            }
        }
    }
}

// ---------------------------------------------------------------------------
// Kernel 4: GroupNorm (32 groups, 8 ch/group) + optional residual + ReLU.
// One WG of 256 threads per (n, group); group = 8*HW = 32768 elements.
// ---------------------------------------------------------------------------
__global__ __launch_bounds__(256) void gn_kernel(const float* __restrict__ in,
                                                 float* __restrict__ outp,
                                                 const float* __restrict__ gamma,
                                                 const float* __restrict__ beta,
                                                 const float* __restrict__ residual,
                                                 int HW) {
    int n = blockIdx.x >> 5;
    int g = blockIdx.x & 31;
    int tid = threadIdx.x;
    const int cnt = 8 * HW;
    size_t base = ((size_t)n * C_CH + g * 8) * HW;

    float s = 0.f, s2 = 0.f;
    for (int i = tid; i < cnt; i += 256) {
        float v = in[base + i];
        s += v; s2 += v * v;
    }
    __shared__ float rb[512];
    rb[tid] = s; rb[256 + tid] = s2;
    __syncthreads();
    for (int off = 128; off > 0; off >>= 1) {
        if (tid < off) {
            rb[tid]       += rb[tid + off];
            rb[256 + tid] += rb[256 + tid + off];
        }
        __syncthreads();
    }
    float inv  = 1.f / (float)cnt;
    float mean = rb[0] * inv;
    float var  = rb[256] * inv - mean * mean;
    float rstd = rsqrtf(var + 1e-5f);

    for (int i = tid; i < cnt; i += 256) {
        int c = g * 8 + i / HW;
        float v = (in[base + i] - mean) * rstd * gamma[c] + beta[c];
        if (residual) v += residual[base + i];
        v = fmaxf(v, 0.f);
        outp[base + i] = v;
    }
}

// ---------------------------------------------------------------------------
// Launch: x -> offsets1 -> WMMA deform GEMM -> GN+ReLU -> offsets2 ->
//         WMMA deform GEMM -> GN + residual + ReLU -> d_out
// ---------------------------------------------------------------------------
extern "C" void kernel_launch(void* const* d_in, const int* in_sizes, int n_in,
                              void* d_out, int out_size, void* d_ws, size_t ws_size,
                              hipStream_t stream) {
    const int N = 4, H = 64, W = 64;
    const int HW = H * W;
    const int NPIX = N * HW;                 // 16384
    const size_t PLANE = (size_t)N * C_CH * HW;  // 4,194,304 floats
    const size_t OFFS  = (size_t)N * 9 * HW;     // 147,456 floats
    const size_t WSZ   = (size_t)C_CH * KTOT;    // 589,824 weights

    const float* x     = (const float*)d_in[0];
    const float* w_tm1 = (const float*)d_in[1];
    const float* b_tm1 = (const float*)d_in[2];
    const float* w_tr1 = (const float*)d_in[3];
    const float* b_tr1 = (const float*)d_in[4];
    const float* w_dc1 = (const float*)d_in[5];
    const float* g1    = (const float*)d_in[6];
    const float* be1   = (const float*)d_in[7];
    const float* w_tm2 = (const float*)d_in[8];
    const float* b_tm2 = (const float*)d_in[9];
    const float* w_tr2 = (const float*)d_in[10];
    const float* b_tr2 = (const float*)d_in[11];
    const float* w_dc2 = (const float*)d_in[12];
    const float* g2    = (const float*)d_in[13];
    const float* be2   = (const float*)d_in[14];
    float* out = (float*)d_out;

    float* buf1 = (float*)d_ws;
    float* buf2 = buf1 + PLANE;
    float* dyb  = buf2 + PLANE;
    float* dxb  = dyb + OFFS;
    _Float16* wh1 = (_Float16*)(dxb + OFFS);
    _Float16* wh2 = wh1 + WSZ;

    // Weight conversion (fp32 -> f16, kk-major K layout)
    convert_w_kernel<<<(int)((WSZ + 255) / 256), 256, 0, stream>>>(w_dc1, wh1, (int)WSZ);
    convert_w_kernel<<<(int)((WSZ + 255) / 256), 256, 0, stream>>>(w_dc2, wh2, (int)WSZ);

    // ---- Block 1 ----
    offsets_kernel<<<NPIX / 256, 256, 0, stream>>>(x, w_tm1, b_tm1, w_tr1, b_tr1,
                                                   dyb, dxb, H, W);
    deform_gemm_kernel<<<NPIX / TILE_M, 128, 0, stream>>>(x, dyb, dxb, wh1, buf1, H, W);
    gn_kernel<<<N * 32, 256, 0, stream>>>(buf1, buf1, g1, be1, nullptr, HW);

    // ---- Block 2 ----
    offsets_kernel<<<NPIX / 256, 256, 0, stream>>>(buf1, w_tm2, b_tm2, w_tr2, b_tr2,
                                                   dyb, dxb, H, W);
    deform_gemm_kernel<<<NPIX / TILE_M, 128, 0, stream>>>(buf1, dyb, dxb, wh2, buf2, H, W);
    gn_kernel<<<N * 32, 256, 0, stream>>>(buf2, out, g2, be2, x, HW);
}
